// MyModel_87454124082138
// MI455X (gfx1250) — matrix operand order repaired and checked by reading, not compiled, not run on hardware
//
#include <hip/hip_runtime.h>
#include <stdint.h>

// ---------------------------------------------------------------------------
// Types / helpers
// ---------------------------------------------------------------------------
typedef __bf16 v16bf __attribute__((ext_vector_type(16)));
typedef __bf16 v2bf  __attribute__((ext_vector_type(2)));
typedef float  v8f   __attribute__((ext_vector_type(8)));

struct alignas(16) U4 { uint32_t x, y, z, w; };
struct alignas(16) F4 { float x, y, z, w; };

union FragBF {
    v16bf    bf;
    uint32_t u[8];
};

#define T_ 256
#define B_ 128
#define D_ 1024
#define G_ 512          // 4*H
#define H_ 128

#define SHS 68          // padded h-row stride in dwords (272 B, 16B-aligned, bank-spread)
#define SH_U32 (16 * SHS)
#define SWR_U32 32768   // Wr swizzled: 32 ntiles * 4 kchunks * 256 dwords = 128 KB
#define LSTM_SMEM_BYTES ((SWR_U32 + SH_U32) * 4)

// Native bf16 conversion path (ISel picks v_cvt_pk_bf16_f32 on gfx1250).
__device__ __forceinline__ uint32_t pk_bf16(float lo, float hi) {
    v2bf v;
    v[0] = (__bf16)lo;
    v[1] = (__bf16)hi;
    return __builtin_bit_cast(uint32_t, v);
}
__device__ __forceinline__ uint16_t bf16c(float f) {
    __bf16 b = (__bf16)f;
    return __builtin_bit_cast(uint16_t, b);
}
__device__ __forceinline__ float bflo(uint32_t u) { return __uint_as_float(u << 16); }
__device__ __forceinline__ float bfhi(uint32_t u) { return __uint_as_float(u & 0xffff0000u); }

__device__ __forceinline__ float sigm(float x) {
    return 1.0f / (1.0f + __expf(-x));
}
__device__ __forceinline__ float tanh_fast(float x) {
    x = fminf(fmaxf(x, -15.0f), 15.0f);
    float e = __expf(2.0f * x);
    return (e - 1.0f) / (e + 1.0f);
}

__device__ __forceinline__ v8f wmma_bf16(const FragBF& a, const FragBF& b, v8f c) {
    return __builtin_amdgcn_wmma_f32_16x16x32_bf16(
        false, a.bf, false, b.bf, (short)0, c, false, false);
}

// B-fragment layout in memory (global ws or LDS), per (ntile, kchunk): 256 dwords.
//   dword[half*128 + lane*4 + c] = pack(W[kb][col], W[kb+1][col])
//   with v = half*4 + c, col = nt*16 + (lane&15), kb = kc*32 + ((lane>=16)?8:0) + 16*(v>>2) + 2*(v&3)
// -> each lane reads two 16B chunks at lane*16B stride: conflict-free b128 loads.
__device__ __forceinline__ void load_bfrag(FragBF& f, const uint32_t* base,
                                           int nt, int KC, int kc, int lane) {
    const uint32_t* p = base + (((nt * KC + kc) << 8) + (lane << 2));
    U4 q0 = *(const U4*)p;
    U4 q1 = *(const U4*)(p + 128);
    f.u[0] = q0.x; f.u[1] = q0.y; f.u[2] = q0.z; f.u[3] = q0.w;
    f.u[4] = q1.x; f.u[5] = q1.y; f.u[6] = q1.z; f.u[7] = q1.w;
}

// A-fragment straight from global f32 rows (convert to bf16 in-register).
// A 16x32 bf16 layout: lane m=lane&15; kb = k0 + ((lane>=16)?8:0);
// VGPR v=0..3 : K = kb+2v,kb+2v+1 ; VGPR v=4..7 : K = kb+16+2(v-4), +1.
__device__ __forceinline__ void load_afrag_g(FragBF& f, const float* base,
                                             long rowStride, int k0, int lane) {
    int m  = lane & 15;
    int kb = k0 + ((lane >> 4) << 3);
    const float* p = base + (long)m * rowStride + kb;
    F4 q0 = *(const F4*)(p);
    F4 q1 = *(const F4*)(p + 4);
    F4 q2 = *(const F4*)(p + 16);
    F4 q3 = *(const F4*)(p + 20);
    f.u[0] = pk_bf16(q0.x, q0.y); f.u[1] = pk_bf16(q0.z, q0.w);
    f.u[2] = pk_bf16(q1.x, q1.y); f.u[3] = pk_bf16(q1.z, q1.w);
    f.u[4] = pk_bf16(q2.x, q2.y); f.u[5] = pk_bf16(q2.z, q2.w);
    f.u[6] = pk_bf16(q3.x, q3.y); f.u[7] = pk_bf16(q3.z, q3.w);
}

// A-fragment from bf16 h tile in LDS ([16][SHS dwords], padded -> bank-conflict-free).
__device__ __forceinline__ void load_afrag_lds(FragBF& f, const uint32_t* sH,
                                               int kc, int lane) {
    int m  = lane & 15;
    int kb = kc * 32 + ((lane >> 4) << 3);
    const uint32_t* p = sH + m * SHS + (kb >> 1);
    U4 q0 = *(const U4*)p;
    U4 q1 = *(const U4*)(p + 8);
    f.u[0] = q0.x; f.u[1] = q0.y; f.u[2] = q0.z; f.u[3] = q0.w;
    f.u[4] = q1.x; f.u[5] = q1.y; f.u[6] = q1.z; f.u[7] = q1.w;
}

// xp is stored in WMMA C-fragment layout (produced & consumed only here):
//   dword index = (((t*8 + bt)*32 + nt)*32 + lane)*4 + c,  c packs (r=2c, r=2c+1)
__device__ __forceinline__ long xp_idx(int t, int bt, int nt, int lane) {
    return ((((long)t * 8 + bt) * 32 + nt) * 32 + lane) << 2;
}

// ---------------------------------------------------------------------------
// Kernel 0: swizzle f32 weight [K][N] -> bf16 WMMA B-fragment layout
// ---------------------------------------------------------------------------
__global__ void swizzle_w(const float* __restrict__ W, uint32_t* __restrict__ dst,
                          int K, int N) {
    int idx = blockIdx.x * blockDim.x + threadIdx.x;
    int NT = N >> 4, KC = K >> 5;
    int total = NT * KC * 256;
    if (idx >= total) return;
    int c    = idx & 3;
    int lane = (idx >> 2) & 31;
    int half = (idx >> 7) & 1;
    int fc   = idx >> 8;
    int kc   = fc % KC;
    int nt   = fc / KC;
    int v    = half * 4 + c;
    int col  = nt * 16 + (lane & 15);
    int kb   = kc * 32 + ((lane >> 4) << 3) + ((v >> 2) << 4) + 2 * (v & 3);
    dst[idx] = pk_bf16(W[(long)kb * N + col], W[(long)(kb + 1) * N + col]);
}

// ---------------------------------------------------------------------------
// Kernel 1: input projection -> xp fragments (bf16) = x(b, t|T-1-t, :) @ Wk + bias
// Grid (4 Ntiles-of-128, T, 2 dirs), 256 threads = 8 waves.
// Wave tile: M=16 (one A-frag per K-chunk -> minimal f32->bf16 conversions)
//            x N=128 (8 tiles). K-loop over 1024 in chunks of 32: 8 wmma/chunk.
// All 8 B fragments live at once: ~17 outstanding b128 loads per wave, wmmas
// retire against staged loadcnt waits instead of draining to zero.
// Epilogue stores each 16x16 tile as one b128 per lane (C-fragment layout).
// ---------------------------------------------------------------------------
__global__ void __launch_bounds__(256) proj_kernel(
    const float* __restrict__ x,
    const uint32_t* __restrict__ swzWk_f, const uint32_t* __restrict__ swzWk_b,
    const float* __restrict__ bias_f, const float* __restrict__ bias_b,
    uint32_t* __restrict__ xp_f, uint32_t* __restrict__ xp_b)
{
    const int dir = blockIdx.z;
    const uint32_t* swz = dir ? swzWk_b : swzWk_f;
    const float*   bias = dir ? bias_b  : bias_f;
    uint32_t*        xp = dir ? xp_b    : xp_f;
    const int t    = blockIdx.y;
    const int tsrc = dir ? (T_ - 1 - t) : t;
    const int n0   = blockIdx.x * 128;
    const int nt0  = n0 >> 4;
    const int lane = threadIdx.x & 31;
    const int w    = threadIdx.x >> 5;   // wave = batch-tile bt (rows 16w..16w+15)

    v8f acc[8];
#pragma unroll
    for (int nn = 0; nn < 8; ++nn) acc[nn] = (v8f)0.0f;

    const long rowStride = (long)T_ * D_;  // floats between batch rows of x
    const float* abase = x + (long)(w * 16) * rowStride + (long)tsrc * D_;

    for (int kc = 0; kc < 32; ++kc) {
        // Issue every load for this K-chunk first (16 B-frag b128 + 4 A b128),
        // then compute: maximizes per-wave memory-level parallelism.
        FragBF bb[8];
#pragma unroll
        for (int nn = 0; nn < 8; ++nn)
            load_bfrag(bb[nn], swz, nt0 + nn, 32, kc, lane);
        FragBF a;
        load_afrag_g(a, abase, rowStride, kc * 32, lane);
#pragma unroll
        for (int nn = 0; nn < 8; ++nn)
            acc[nn] = wmma_bf16(a, bb[nn], acc[nn]);
    }

    // Epilogue: add bias, pack to bf16 pairs, one b128 store per tile per lane.
    const int nl = lane & 15;
#pragma unroll
    for (int nn = 0; nn < 8; ++nn) {
        int nt  = nt0 + nn;
        float bc = bias[nt * 16 + nl];
        U4 st;
        st.x = pk_bf16(acc[nn][0] + bc, acc[nn][1] + bc);
        st.y = pk_bf16(acc[nn][2] + bc, acc[nn][3] + bc);
        st.z = pk_bf16(acc[nn][4] + bc, acc[nn][5] + bc);
        st.w = pk_bf16(acc[nn][6] + bc, acc[nn][7] + bc);
        *(U4*)(xp + xp_idx(t, w, nt, lane)) = st;
    }
}

// ---------------------------------------------------------------------------
// Kernel 2: recurrent LSTM. Batch rows are independent -> 8 WGs x 16 rows per
// direction, zero cross-WG sync. Wr staged into LDS (128 KB) with gfx1250
// async-DMA loads (no VGPR round-trip, tracked by ASYNCcnt); h tile (16x128
// bf16, padded) in LDS; cell state c in 8 VGPRs/lane for all 256 steps.
// Wave w owns gate-column block [16w,16w+16) across i/f/g/o (N-tiles w,w+8,w+16,w+24).
// xp arrives in C-fragment layout: one global_load_b128 per gate per step.
// Grid (8, 2 dirs), 256 threads = 8 waves. 128 wmma per WG per timestep.
// ---------------------------------------------------------------------------
__global__ void __launch_bounds__(256) lstm_kernel(
    const uint32_t* __restrict__ xp_f, const uint32_t* __restrict__ xp_b,
    const uint32_t* __restrict__ swzWr_f, const uint32_t* __restrict__ swzWr_b,
    const float* __restrict__ bias_f, const float* __restrict__ bias_b,
    float* __restrict__ hall)
{
    extern __shared__ uint32_t smem[];
    uint32_t* sWr = smem;              // 32768 dwords = 128 KB
    uint32_t* sH  = smem + SWR_U32;    // 16*SHS dwords (h tile, bf16, padded rows)

    const int dir = blockIdx.y;
    const uint32_t* xp  = dir ? xp_b    : xp_f;
    const uint32_t* swz = dir ? swzWr_b : swzWr_f;
    const float*   bias = dir ? bias_b  : bias_f;
    const int bt   = blockIdx.x;       // batch tile (rows 16*bt..16*bt+15)
    const int tid  = threadIdx.x;
    const int lane = tid & 31;
    const int w    = tid >> 5;

    // Stage swizzled Wr into LDS via async DMA (GLOBAL_LOAD_ASYNC_TO_LDS_B128,
    // ASYNCcnt-tracked). Generic shared pointer's low 32 bits are the LDS byte
    // address (aperture rule, ISA 10.2). Zero-init of h overlaps the DMA.
    {
        const uint32_t ldsbase = (uint32_t)(uintptr_t)sWr;
        for (int i = tid * 4; i < SWR_U32; i += 256 * 4) {
            uint32_t ldsaddr = ldsbase + (uint32_t)i * 4u;
            unsigned long long ga = (unsigned long long)(uintptr_t)(swz + i);
            asm volatile("global_load_async_to_lds_b128 %0, %1, off"
                         :: "v"(ldsaddr), "v"(ga) : "memory");
        }
    }
    for (int i = tid; i < SH_U32; i += 256) sH[i] = 0u;
    asm volatile("s_wait_asynccnt 0x0" ::: "memory");
    __syncthreads();

    const int nl   = lane & 15;
    const int mh   = (lane >> 4) << 3;
    const int jcol = w * 16 + nl;              // gate-column within H
    float bgate[4];
#pragma unroll
    for (int g = 0; g < 4; ++g) bgate[g] = bias[g * H_ + jcol];

    float creg[8], hreg[8];
#pragma unroll
    for (int r = 0; r < 8; ++r) { creg[r] = 0.0f; hreg[r] = 0.0f; }

    uint16_t* sH16 = (uint16_t*)sH;

    for (int t = 0; t < T_; ++t) {
        if (t + 1 < T_)
            __builtin_prefetch((const void*)(xp + xp_idx(t + 1, bt, w, lane)), 0, 0);

        // Accumulators initialized with xp(t) + bias (already in C-frag layout).
        v8f acc[4];
#pragma unroll
        for (int g = 0; g < 4; ++g) {
            U4 q = *(const U4*)(xp + xp_idx(t, bt, g * 8 + w, lane));
            float bg = bgate[g];
            acc[g][0] = bflo(q.x) + bg; acc[g][1] = bfhi(q.x) + bg;
            acc[g][2] = bflo(q.y) + bg; acc[g][3] = bfhi(q.y) + bg;
            acc[g][4] = bflo(q.z) + bg; acc[g][5] = bfhi(q.z) + bg;
            acc[g][6] = bflo(q.w) + bg; acc[g][7] = bfhi(q.w) + bg;
        }

        // z += h @ Wr  (K = 128 in 4 chunks of 32, bf16 WMMA, f32 accumulate).
        // Per chunk: issue A + all 4 B fragment loads, then 4 independent wmmas.
#pragma unroll
        for (int kc = 0; kc < 4; ++kc) {
            FragBF bfr[4];
#pragma unroll
            for (int g = 0; g < 4; ++g)
                load_bfrag(bfr[g], sWr, g * 8 + w, 4, kc, lane);
            FragBF a;
            load_afrag_lds(a, sH, kc, lane);
#pragma unroll
            for (int g = 0; g < 4; ++g)
                acc[g] = wmma_bf16(a, bfr[g], acc[g]);
        }

        // Gates + state update entirely in registers (keras order i,f,g,o).
#pragma unroll
        for (int r = 0; r < 8; ++r) {
            float iv = sigm(acc[0][r]);
            float fv = sigm(acc[1][r]);
            float gv = tanh_fast(acc[2][r]);
            float ov = sigm(acc[3][r]);
            creg[r] = fv * creg[r] + iv * gv;
            hreg[r] = ov * tanh_fast(creg[r]);
        }

        __syncthreads();   // all A-fragment reads of old h done
#pragma unroll
        for (int r = 0; r < 8; ++r)
            sH16[(mh + r) * (SHS * 2) + jcol] = bf16c(hreg[r]);
        __syncthreads();   // new h visible to all waves
    }

    // Final hidden state -> hall[b][dir*128 + j] (f32)
#pragma unroll
    for (int r = 0; r < 8; ++r)
        hall[(long)(bt * 16 + mh + r) * 256 + dir * H_ + jcol] = hreg[r];
}

// ---------------------------------------------------------------------------
// Kernel 3: tiny head MLP. One wave per batch row.
// ---------------------------------------------------------------------------
__global__ void __launch_bounds__(32) head_kernel(
    const float* __restrict__ hall,
    const float* __restrict__ W1, const float* __restrict__ b1,
    const float* __restrict__ W2, const float* __restrict__ b2,
    float* __restrict__ out)
{
    const int b = blockIdx.x;
    const int u = threadIdx.x;          // 0..31 hidden unit
    __shared__ float sy[32];
    float acc = b1[u];
    const float* hb = hall + (long)b * 256;
    for (int d = 0; d < 256; ++d)
        acc = fmaf(hb[d], W1[d * 32 + u], acc);
    sy[u] = fmaxf(acc, 0.0f);
    __syncthreads();
    if (u < 2) {
        float a2 = b2[u];
#pragma unroll
        for (int j = 0; j < 32; ++j)
            a2 = fmaf(sy[j], W2[j * 2 + u], a2);
        out[(long)b * 2 + u] = 1.0f / (1.0f + __expf(-a2));
    }
}

// ---------------------------------------------------------------------------
// Host launcher
// ---------------------------------------------------------------------------
extern "C" void kernel_launch(void* const* d_in, const int* in_sizes, int n_in,
                              void* d_out, int out_size, void* d_ws, size_t ws_size,
                              hipStream_t stream) {
    (void)in_sizes; (void)n_in; (void)out_size; (void)ws_size;
    const float* x   = (const float*)d_in[0];
    const float* Wfk = (const float*)d_in[1];
    const float* Wfr = (const float*)d_in[2];
    const float* bfv = (const float*)d_in[3];
    const float* Wbk = (const float*)d_in[4];
    const float* Wbr = (const float*)d_in[5];
    const float* bbv = (const float*)d_in[6];
    const float* W1  = (const float*)d_in[7];
    const float* b1  = (const float*)d_in[8];
    const float* W2  = (const float*)d_in[9];
    const float* b2  = (const float*)d_in[10];

    char* ws = (char*)d_ws;
    uint32_t* swzWfk = (uint32_t*)(ws);                                   // 1 MB
    uint32_t* swzWbk = (uint32_t*)(ws + ((size_t)1 << 20));               // 1 MB
    uint32_t* swzWfr = (uint32_t*)(ws + ((size_t)2 << 20));               // 128 KB
    uint32_t* swzWbr = (uint32_t*)(ws + ((size_t)2 << 20) + (128u << 10));// 128 KB
    uint32_t* xp_f   = (uint32_t*)(ws + ((size_t)4 << 20));               // 32 MB
    uint32_t* xp_b   = (uint32_t*)(ws + ((size_t)36 << 20));              // 32 MB
    float*    hall   = (float*)   (ws + ((size_t)68 << 20));              // 128 KB

    // 0) Swizzle weights into WMMA B-fragment bf16 layouts.
    swizzle_w<<<dim3(1024), 256, 0, stream>>>(Wfk, swzWfk, D_, G_);
    swizzle_w<<<dim3(1024), 256, 0, stream>>>(Wbk, swzWbk, D_, G_);
    swizzle_w<<<dim3(128),  256, 0, stream>>>(Wfr, swzWfr, H_, G_);
    swizzle_w<<<dim3(128),  256, 0, stream>>>(Wbr, swzWbr, H_, G_);

    // 1) Input projection (both directions; bwd uses reversed time).
    proj_kernel<<<dim3(4, T_, 2), 256, 0, stream>>>(
        x, swzWfk, swzWbk, bfv, bbv, xp_f, xp_b);

    // 2) Recurrent LSTM, batch-parallel, Wr resident in LDS (135 KB dyn. LDS).
    (void)hipFuncSetAttribute(reinterpret_cast<const void*>(lstm_kernel),
                              hipFuncAttributeMaxDynamicSharedMemorySize,
                              LSTM_SMEM_BYTES);
    lstm_kernel<<<dim3(8, 2), 256, LSTM_SMEM_BYTES, stream>>>(
        xp_f, xp_b, swzWfr, swzWbr, bfv, bbv, hall);

    // 3) Head MLP -> sigmoid probabilities.
    head_kernel<<<dim3(B_), 32, 0, stream>>>(hall, W1, b1, W2, b2, (float*)d_out);
}
